// GraphPredictor_17377437680252
// MI455X (gfx1250) — matrix operand order repaired and checked by reading, not compiled, not run on hardware
//
#include <hip/hip_runtime.h>
#include <hip/hip_bf16.h>

#define N_NODES 8192
#define GNN_DIM 512
#define N_FEAT  64
#define N_ITER  4

typedef __attribute__((ext_vector_type(16))) __bf16 bf16x16;
typedef __attribute__((ext_vector_type(8)))  __bf16 bf16x8;
typedef __attribute__((ext_vector_type(8)))  float  f32x8;

__device__ __forceinline__ f32x8 wmma_bf16(bf16x16 a, bf16x16 b, f32x8 c) {
    return __builtin_amdgcn_wmma_f32_16x16x32_bf16(
        /*neg_a=*/false, a, /*neg_b=*/false, b,
        /*c_mod=*/(short)0, c, /*reuse_a=*/false, /*reuse_b=*/false);
}

__device__ __forceinline__ bf16x16 cat8(bf16x8 lo, bf16x8 hi) {
    return __builtin_shufflevector(lo, hi, 0,1,2,3,4,5,6,7,8,9,10,11,12,13,14,15);
}

// CDNA5 async global->LDS copy (16B per lane), tracked by ASYNCcnt.
__device__ __forceinline__ void async_ld16(unsigned ldsOff, const void* gptr) {
    asm volatile("global_load_async_to_lds_b128 %0, %1, off"
                 :: "v"(ldsOff), "v"((unsigned long long)(uintptr_t)gptr)
                 : "memory");
}
template <int Nw>
__device__ __forceinline__ void wait_async() {
    asm volatile("s_wait_asynccnt %0" :: "n"(Nw) : "memory");
}

// ---------------- conversion kernels ----------------
__global__ void k_f2bf(const float* __restrict__ s, __bf16* __restrict__ d, int n) {
    int i = blockIdx.x * blockDim.x + threadIdx.x;
    if (i < n) d[i] = (__bf16)s[i];
}

// labels [N][D] -> xT [D][N] bf16
__global__ void k_trans_f2bf(const float* __restrict__ s, __bf16* __restrict__ dT) {
    int i = blockIdx.x * blockDim.x + threadIdx.x;
    if (i < N_NODES * GNN_DIM) {
        int n = i >> 9;
        int dd = i & (GNN_DIM - 1);
        dT[(size_t)dd * N_NODES + n] = (__bf16)s[i];
    }
}

// ---------------- GEMM 1: acc = adj @ x  (LDS-staged, async double-buffered) ------
// adjB: [N][N] bf16 row-major (A), xT: [D][N] bf16 (B, K contiguous), acc: [N][D].
// Block: 256 thr = 8 waves, tile 128x128. Wave = 32 rows x 64 cols (2x4 WMMA tiles).
// K chunk = 64: A stage 128x64, B stage 128x64, row stride padded to 144B.
#define KCH       64
#define ROWB      144                 // 128B data + 16B pad (bank spread)
#define BUFB      (128 * ROWB)        // 18432 B per operand
#define BBASE     BUFB                // B region follows A within a buffer
#define STAGEB    (2 * BUFB)          // 36864 B per double-buffer slot

__global__ __launch_bounds__(256) void k_gemm_acc(
    const __bf16* __restrict__ adjB,
    const __bf16* __restrict__ xT,
    __bf16* __restrict__ acc)
{
    __shared__ __align__(16) char smem[2 * STAGEB];   // 73728 B

    // Escape the LDS object's address: the async-copy inline asm writes it, but
    // only receives an integer offset. Without this capture, LLVM proves smem is
    // never stored and folds the ds_loads feeding the WMMAs to undef.
    asm volatile("" :: "v"((void*)smem) : "memory");

    const int tid  = threadIdx.x;
    const int lane = tid & 31;
    const int w    = tid >> 5;
    const int wm   = w & 3;           // 4 M sub-tiles of 32 rows
    const int wn   = w >> 2;          // 2 N sub-tiles of 64 cols
    const int rBase = blockIdx.x * 128;
    const int cBase = blockIdx.y * 128;
    const int m  = lane & 15;
    const int hi = lane >> 4;

    // staging assignment: thread t copies 64B of A row (t>>1) and 64B of B col (t>>1)
    const int sr   = tid >> 1;
    const int shalf = tid & 1;        // which 64B half of the 128B row
    const __bf16* gA = adjB + (size_t)(rBase + sr) * N_NODES + shalf * 32;
    const __bf16* gB = xT   + (size_t)(cBase + sr) * N_NODES + shalf * 32;
    const unsigned lA = (unsigned)(sr * ROWB + shalf * 64);
    const unsigned lB = (unsigned)(BBASE + sr * ROWB + shalf * 64);

    f32x8 c[2][4];
    #pragma unroll
    for (int i = 0; i < 2; ++i)
        #pragma unroll
        for (int j = 0; j < 4; ++j) c[i][j] = f32x8{};

    const int steps = N_NODES / KCH;  // 128

    // prologue: stage chunk 0 into buffer 0
    {
        const __bf16* ga = gA;  const __bf16* gb = gB;
        #pragma unroll
        for (int q = 0; q < 4; ++q) {
            async_ld16(lA + q * 16, ga + q * 8);
            async_ld16(lB + q * 16, gb + q * 8);
        }
    }

    for (int s = 0; s < steps; ++s) {
        const int cur = s & 1;
        __syncthreads();              // prior reads of 'next' buffer complete WG-wide
        if (s + 1 < steps) {
            const unsigned nb = (unsigned)(((s + 1) & 1) * STAGEB);
            const __bf16* ga = gA + (size_t)(s + 1) * KCH;
            const __bf16* gb = gB + (size_t)(s + 1) * KCH;
            #pragma unroll
            for (int q = 0; q < 4; ++q) {
                async_ld16(nb + lA + q * 16, ga + q * 8);
                async_ld16(nb + lB + q * 16, gb + q * 8);
            }
            wait_async<8>();          // current buffer's 8 ops retired (in-order)
        } else {
            wait_async<0>();
        }
        __syncthreads();              // all waves' current-buffer data visible

        const unsigned base = (unsigned)(cur * STAGEB);
        #pragma unroll
        for (int kk = 0; kk < 2; ++kk) {          // two 32-wide K sub-steps
            bf16x16 A[2], B[4];
            #pragma unroll
            for (int i = 0; i < 2; ++i) {
                const char* p = smem + base + (wm * 32 + i * 16 + m) * ROWB
                              + kk * 64 + hi * 16;
                A[i] = cat8(*(const bf16x8*)p, *(const bf16x8*)(p + 32));
            }
            #pragma unroll
            for (int j = 0; j < 4; ++j) {
                const char* q = smem + base + BBASE + (wn * 64 + j * 16 + m) * ROWB
                              + kk * 64 + hi * 16;
                B[j] = cat8(*(const bf16x8*)q, *(const bf16x8*)(q + 32));
            }
            #pragma unroll
            for (int i = 0; i < 2; ++i)
                #pragma unroll
                for (int j = 0; j < 4; ++j)
                    c[i][j] = wmma_bf16(A[i], B[j], c[i][j]);
        }
    }

    // C layout: VGPR r -> row (hi?8:0)+r, col = lane&15
    #pragma unroll
    for (int i = 0; i < 2; ++i) {
        #pragma unroll
        for (int j = 0; j < 4; ++j) {
            const int col = cBase + wn * 64 + j * 16 + m;
            #pragma unroll
            for (int r = 0; r < 8; ++r) {
                const int row = rBase + wm * 32 + i * 16 + hi * 8 + r;
                acc[(size_t)row * GNN_DIM + col] = (__bf16)c[i][j][r];
            }
        }
    }
}

// ---------------- GEMM 2 (fused): xNew = relu(x@Ws^T + b + acc@Wn^T) -------------
__global__ __launch_bounds__(256) void k_update(
    const __bf16* __restrict__ x,
    const __bf16* __restrict__ acc,
    const __bf16* __restrict__ Ws,
    const __bf16* __restrict__ Wn,
    const float*  __restrict__ bself,
    __bf16* __restrict__ xNew,
    __bf16* __restrict__ xT)
{
    const int lane = threadIdx.x & 31;
    const int w    = threadIdx.x >> 5;
    const int wm   = w & 3;
    const int wn   = w >> 2;
    const int rBase = blockIdx.x * 128 + wm * 32;
    const int cBase = blockIdx.y * 64  + wn * 32;
    const int m  = lane & 15;
    const int hi = lane >> 4;
    const int aoff = hi * 16;
    const int boff = hi * 16;

    f32x8 c00 = {}, c01 = {}, c10 = {}, c11 = {};

    const __bf16* xRow0 = x   + (size_t)(rBase + m) * GNN_DIM;
    const __bf16* xRow1 = xRow0 + (size_t)16 * GNN_DIM;
    const __bf16* gRow0 = acc + (size_t)(rBase + m) * GNN_DIM;
    const __bf16* gRow1 = gRow0 + (size_t)16 * GNN_DIM;
    const __bf16* wsCol0 = Ws + (size_t)(cBase + m) * GNN_DIM;
    const __bf16* wsCol1 = wsCol0 + (size_t)16 * GNN_DIM;
    const __bf16* wnCol0 = Wn + (size_t)(cBase + m) * GNN_DIM;
    const __bf16* wnCol1 = wnCol0 + (size_t)16 * GNN_DIM;

    for (int k = 0; k < GNN_DIM; k += 32) {
        bf16x16 a0 = cat8(*(const bf16x8*)(xRow0 + k + aoff),
                          *(const bf16x8*)(xRow0 + k + 8 + aoff));
        bf16x16 a1 = cat8(*(const bf16x8*)(xRow1 + k + aoff),
                          *(const bf16x8*)(xRow1 + k + 8 + aoff));
        bf16x16 b0 = *(const bf16x16*)(wsCol0 + k + boff);
        bf16x16 b1 = *(const bf16x16*)(wsCol1 + k + boff);
        c00 = wmma_bf16(a0, b0, c00);
        c01 = wmma_bf16(a0, b1, c01);
        c10 = wmma_bf16(a1, b0, c10);
        c11 = wmma_bf16(a1, b1, c11);

        bf16x16 g0 = cat8(*(const bf16x8*)(gRow0 + k + aoff),
                          *(const bf16x8*)(gRow0 + k + 8 + aoff));
        bf16x16 g1 = cat8(*(const bf16x8*)(gRow1 + k + aoff),
                          *(const bf16x8*)(gRow1 + k + 8 + aoff));
        bf16x16 h0 = *(const bf16x16*)(wnCol0 + k + boff);
        bf16x16 h1 = *(const bf16x16*)(wnCol1 + k + boff);
        c00 = wmma_bf16(g0, h0, c00);
        c01 = wmma_bf16(g0, h1, c01);
        c10 = wmma_bf16(g1, h0, c10);
        c11 = wmma_bf16(g1, h1, c11);
    }

    const int col0 = cBase + m;
    const int col1 = col0 + 16;
    const float bias0 = bself[col0];
    const float bias1 = bself[col1];
    #pragma unroll
    for (int r = 0; r < 8; ++r) {
        int row0 = rBase + hi * 8 + r;
        int row1 = row0 + 16;
        float v00 = c00[r] + bias0; v00 = v00 > 0.f ? v00 : 0.f;
        float v01 = c01[r] + bias1; v01 = v01 > 0.f ? v01 : 0.f;
        float v10 = c10[r] + bias0; v10 = v10 > 0.f ? v10 : 0.f;
        float v11 = c11[r] + bias1; v11 = v11 > 0.f ? v11 : 0.f;
        xNew[(size_t)row0 * GNN_DIM + col0] = (__bf16)v00;
        xNew[(size_t)row0 * GNN_DIM + col1] = (__bf16)v01;
        xNew[(size_t)row1 * GNN_DIM + col0] = (__bf16)v10;
        xNew[(size_t)row1 * GNN_DIM + col1] = (__bf16)v11;
        xT[(size_t)col0 * N_NODES + row0] = (__bf16)v00;
        xT[(size_t)col1 * N_NODES + row0] = (__bf16)v01;
        xT[(size_t)col0 * N_NODES + row1] = (__bf16)v10;
        xT[(size_t)col1 * N_NODES + row1] = (__bf16)v11;
    }
}

// ---------------- pooling + FC + sigmoid ----------------
__global__ __launch_bounds__(256) void k_pool_partial(const __bf16* __restrict__ x,
                                                      float* __restrict__ part) {
    const int b = blockIdx.x;
    const int rBase = b * 128;
    for (int d = threadIdx.x; d < GNN_DIM; d += 256) {
        float s = 0.f;
        for (int r = 0; r < 128; ++r)
            s += (float)x[(size_t)(rBase + r) * GNN_DIM + d];
        part[b * GNN_DIM + d] = s;
    }
}

__global__ __launch_bounds__(512) void k_pool_fc(const float* __restrict__ part,
                                                 const float* __restrict__ feat,
                                                 const float* __restrict__ Wfc,
                                                 const float* __restrict__ bfc,
                                                 float* __restrict__ out) {
    __shared__ float red[512];
    const int d = threadIdx.x;
    float s = 0.f;
    for (int b = 0; b < 64; ++b) s += part[b * GNN_DIM + d];
    float pooled = s * (1.0f / (float)N_NODES);
    float p = pooled * Wfc[d];
    if (d < N_FEAT) p += feat[d] * Wfc[GNN_DIM + d];
    red[d] = p;
    __syncthreads();
    for (int off = 256; off > 0; off >>= 1) {
        if (d < off) red[d] += red[d + off];
        __syncthreads();
    }
    if (d == 0) {
        float z = red[0] + bfc[0];
        out[0] = 1.0f / (1.0f + expf(-z));
    }
}

// ---------------- host ----------------
extern "C" void kernel_launch(void* const* d_in, const int* in_sizes, int n_in,
                              void* d_out, int out_size, void* d_ws, size_t ws_size,
                              hipStream_t stream) {
    const float* labels   = (const float*)d_in[0];
    const float* adj      = (const float*)d_in[1];
    const float* features = (const float*)d_in[2];
    const float* W_self   = (const float*)d_in[3];
    const float* b_self   = (const float*)d_in[4];
    const float* W_neigh  = (const float*)d_in[5];
    const float* W_fc     = (const float*)d_in[6];
    const float* b_fc     = (const float*)d_in[7];
    float* out = (float*)d_out;

    char* ws = (char*)d_ws;
    size_t off = 0;
    auto alloc = [&](size_t bytes) -> void* {
        void* p = ws + off;
        off += (bytes + 255) & ~(size_t)255;
        return p;
    };
    __bf16* adjB = (__bf16*)alloc((size_t)N_NODES * N_NODES * 2);
    __bf16* x0   = (__bf16*)alloc((size_t)N_NODES * GNN_DIM * 2);
    __bf16* x1   = (__bf16*)alloc((size_t)N_NODES * GNN_DIM * 2);
    __bf16* xT   = (__bf16*)alloc((size_t)GNN_DIM * N_NODES * 2);
    __bf16* acc  = (__bf16*)alloc((size_t)N_NODES * GNN_DIM * 2);
    __bf16* WsB  = (__bf16*)alloc((size_t)GNN_DIM * GNN_DIM * 2);
    __bf16* WnB  = (__bf16*)alloc((size_t)GNN_DIM * GNN_DIM * 2);
    float*  part = (float*)alloc((size_t)64 * GNN_DIM * 4);

    {
        int n = N_NODES * N_NODES;
        k_f2bf<<<(n + 255) / 256, 256, 0, stream>>>(adj, adjB, n);
        n = N_NODES * GNN_DIM;
        k_f2bf<<<(n + 255) / 256, 256, 0, stream>>>(labels, x0, n);
        k_trans_f2bf<<<(n + 255) / 256, 256, 0, stream>>>(labels, xT);
        n = GNN_DIM * GNN_DIM;
        k_f2bf<<<(n + 255) / 256, 256, 0, stream>>>(W_self, WsB, n);
        k_f2bf<<<(n + 255) / 256, 256, 0, stream>>>(W_neigh, WnB, n);
    }

    __bf16* xc = x0;
    __bf16* xn = x1;
    dim3 g1(N_NODES / 128, GNN_DIM / 128);   // 64 x 4
    dim3 g2(N_NODES / 128, GNN_DIM / 64);    // 64 x 8
    for (int it = 0; it < N_ITER; ++it) {
        k_gemm_acc<<<g1, 256, 0, stream>>>(adjB, xT, acc);
        k_update<<<g2, 256, 0, stream>>>(xc, acc, WsB, WnB, b_self, xn, xT);
        __bf16* t = xc; xc = xn; xn = t;
    }

    k_pool_partial<<<64, 256, 0, stream>>>(xc, part);
    k_pool_fc<<<1, 512, 0, stream>>>(part, features, W_fc, b_fc, out);
}
